// BBoxHead_13692355740313
// MI455X (gfx1250) — matrix operand order repaired and checked by reading, not compiled, not run on hardware
//
#include <hip/hip_runtime.h>
#include <hip/hip_bf16.h>

// BBoxHead fused cls+reg FC: out[8192 x 405] = feat[8192 x 12544] @ W_pad^T + bias
// bf16x3 split-precision GEMM on v_wmma_f32_16x16x32_bf16 (wave32).
//   a = a_hi + a_lo (bf16 truncation + exact residual), same for b;
//   result = a_hi*b_hi + a_hi*b_lo + a_lo*b_hi, each term in its OWN fp32
//   accumulator (12 independent WMMAs per K-step, no hazard NOPs).
// Weights are pre-split into d_ws in FRAGMENT-MAJOR order so the GEMM's B
// stream is one sequential pointer with immediate offsets (no per-tile
// pointer arithmetic near the WMMAs -> no WMMA->VALU WAR hazard NOPs).

#define KDIM  12544
#define NCLS  81
#define NREG  324
#define NOUT  405    // 81 + 324
#define NPAD  448    // 7 groups of 64 columns
#define NGRP  7      // column groups
#define KBLK  392    // KDIM / 32
// Bpack element index: (((kb*NGRP + g)*4 + t)*32 + lane)*16 + e
//   col = g*64 + t*16 + (lane&15),  k = kb*32 + (e>>3)*16 + (lane>>4)*8 + (e&7)
#define KB_STRIDE (NGRP * 4 * 32 * 16)   // 14336 elements per kb step
#define G_STRIDE  (4 * 32 * 16)          // 2048 elements per column group
#define T_STRIDE  (32 * 16)              // 512 elements per tile

typedef __attribute__((ext_vector_type(16))) __bf16 v16bf;
typedef __attribute__((ext_vector_type(8)))  float  v8f;

union BF16Frag {
    unsigned int u[8];
    v16bf        v;
};

// pack two truncated-bf16 values (element 2i in low half of dword i)
__device__ __forceinline__ unsigned int pack_bf2(unsigned int u0, unsigned int u1) {
    return (u0 >> 16) | (u1 & 0xFFFF0000u);
}

// split two fp32 into packed bf16 hi dword and packed bf16 lo (residual) dword
__device__ __forceinline__ void split2(float f0, float f1,
                                       unsigned int& hi, unsigned int& lo) {
    unsigned int u0 = __float_as_uint(f0);
    unsigned int u1 = __float_as_uint(f1);
    hi = pack_bf2(u0, u1);
    float l0 = f0 - __uint_as_float(u0 & 0xFFFF0000u);  // exact residual
    float l1 = f1 - __uint_as_float(u1 & 0xFFFF0000u);
    lo = pack_bf2(__float_as_uint(l0), __float_as_uint(l1));
}

// ---------------------------------------------------------------------------
// Prep: split W_cls/W_reg into fragment-major bf16 hi/lo Bpack arrays and the
// padded bias vector. Columns >= NOUT are zero.
// ---------------------------------------------------------------------------
__global__ void prep_weights(const float* __restrict__ Wc, const float* __restrict__ bc,
                             const float* __restrict__ Wr, const float* __restrict__ br,
                             unsigned short* __restrict__ Bhi,
                             unsigned short* __restrict__ Blo,
                             float* __restrict__ bias) {
    size_t i = (size_t)blockIdx.x * blockDim.x + threadIdx.x;
    const size_t total = (size_t)NPAD * KDIM;
    if (i < total) {
        const int    e    = (int)(i & 15);
        const int    lane = (int)((i >> 4) & 31);
        const int    t    = (int)((i >> 9) & 3);
        const size_t rest = i >> 11;
        const int    g    = (int)(rest % NGRP);
        const int    kb   = (int)(rest / NGRP);
        const int    col  = g * 64 + t * 16 + (lane & 15);
        const int    k    = kb * 32 + (e >> 3) * 16 + (lane >> 4) * 8 + (e & 7);
        float w = 0.0f;
        if (col < NCLS)      w = Wc[(size_t)col * KDIM + k];
        else if (col < NOUT) w = Wr[(size_t)(col - NCLS) * KDIM + k];
        unsigned int u = __float_as_uint(w);
        float lo = w - __uint_as_float(u & 0xFFFF0000u);
        Bhi[i] = (unsigned short)(u >> 16);
        Blo[i] = (unsigned short)(__float_as_uint(lo) >> 16);
    }
    if (i < NPAD) {
        float b = 0.0f;
        if (i < NCLS)      b = bc[i];
        else if (i < NOUT) b = br[i - NCLS];
        bias[i] = b;
    }
}

// ---------------------------------------------------------------------------
// GEMM: one wave computes a 16x64 output tile (4 N-tiles), K-loop step 32.
// grid = (NGRP, nRows/16), block = 32 threads (one wave32).
// ---------------------------------------------------------------------------
__global__ __launch_bounds__(32) void bbox_wmma_gemm(
    const float* __restrict__ x,
    const unsigned short* __restrict__ Bhi,
    const unsigned short* __restrict__ Blo,
    const float* __restrict__ bias,
    float* __restrict__ out, int nRows)
{
    const int lane  = threadIdx.x;      // 0..31
    const int sel   = lane >> 4;        // lane group 0 / 1
    const int lmod  = lane & 15;
    const int g     = blockIdx.x;       // column group (64 cols)
    const int mBase = blockIdx.y << 4;  // 16-row output panel
    const int nBase = g << 6;

    // A fragment (16x32 bf16, ISA layout): this lane supplies row (mBase+lmod),
    // K offsets {sel*8 + 0..7} and {16 + sel*8 + 0..7} within each K-step.
    const float* aPtr = x + (size_t)(mBase + lmod) * KDIM + (sel << 3);

    // Fragment-major B stream: single sequential pointer per hi/lo; all four
    // tiles are reached with constant immediate offsets (t*1024 bytes).
    const unsigned short* bhPtr = Bhi + (size_t)g * G_STRIDE + (size_t)lane * 16;
    const unsigned short* blPtr = Blo + (size_t)g * G_STRIDE + (size_t)lane * 16;

    // One accumulator per split term per tile -> all 12 WMMAs per iteration
    // are independent; carried dependency distance >= 12 (no hazard NOPs).
    v8f accHH[4] = {};
    v8f accHL[4] = {};
    v8f accLH[4] = {};

#pragma unroll 2
    for (int kb = 0; kb < KBLK; ++kb) {
        // stream A ahead (speculative prefetch -> global_prefetch_b8)
        __builtin_prefetch(aPtr + 128, 0, 1);

        // this lane's 16 A floats (two 32B chunks, b128 loads, imm offsets)
        const float4 fa0 = *(const float4*)(aPtr);
        const float4 fa1 = *(const float4*)(aPtr + 4);
        const float4 fa2 = *(const float4*)(aPtr + 16);
        const float4 fa3 = *(const float4*)(aPtr + 20);

        BF16Frag ah, al;
        split2(fa0.x, fa0.y, ah.u[0], al.u[0]);
        split2(fa0.z, fa0.w, ah.u[1], al.u[1]);
        split2(fa1.x, fa1.y, ah.u[2], al.u[2]);
        split2(fa1.z, fa1.w, ah.u[3], al.u[3]);
        split2(fa2.x, fa2.y, ah.u[4], al.u[4]);
        split2(fa2.z, fa2.w, ah.u[5], al.u[5]);
        split2(fa3.x, fa3.y, ah.u[6], al.u[6]);
        split2(fa3.z, fa3.w, ah.u[7], al.u[7]);

#pragma unroll
        for (int t = 0; t < 4; ++t) {
            const uint4 h0 = *(const uint4*)(bhPtr + t * T_STRIDE);
            const uint4 h1 = *(const uint4*)(bhPtr + t * T_STRIDE + 8);
            const uint4 l0 = *(const uint4*)(blPtr + t * T_STRIDE);
            const uint4 l1 = *(const uint4*)(blPtr + t * T_STRIDE + 8);
            BF16Frag bh, bl;
            bh.u[0] = h0.x; bh.u[1] = h0.y; bh.u[2] = h0.z; bh.u[3] = h0.w;
            bh.u[4] = h1.x; bh.u[5] = h1.y; bh.u[6] = h1.z; bh.u[7] = h1.w;
            bl.u[0] = l0.x; bl.u[1] = l0.y; bl.u[2] = l0.z; bl.u[3] = l0.w;
            bl.u[4] = l1.x; bl.u[5] = l1.y; bl.u[6] = l1.z; bl.u[7] = l1.w;

            accHH[t] = __builtin_amdgcn_wmma_f32_16x16x32_bf16(
                false, ah.v, false, bh.v, (short)0, accHH[t], false, false);
            accHL[t] = __builtin_amdgcn_wmma_f32_16x16x32_bf16(
                false, ah.v, false, bl.v, (short)0, accHL[t], false, false);
            accLH[t] = __builtin_amdgcn_wmma_f32_16x16x32_bf16(
                false, al.v, false, bh.v, (short)0, accLH[t], false, false);
        }

        aPtr  += 32;
        bhPtr += KB_STRIDE;
        blPtr += KB_STRIDE;
    }

    // Epilogue. C/D layout: VGPR r, lane l -> M = r + 8*(l>>4), N = l&15.
    const size_t regBlock = (size_t)nRows * NCLS;  // start of bbox_pred in d_out
#pragma unroll
    for (int t = 0; t < 4; ++t) {
        const int col = nBase + t * 16 + lmod;
        if (col < NOUT) {
            const float bv = bias[col];
#pragma unroll
            for (int r = 0; r < 8; ++r) {
                const int row = mBase + r + (sel << 3);
                const float v = ((accHH[t][r] + accHL[t][r]) + accLH[t][r]) + bv;
                if (col < NCLS)
                    out[(size_t)row * NCLS + col] = v;
                else
                    out[regBlock + (size_t)row * NREG + (col - NCLS)] = v;
            }
        }
    }
}

// ---------------------------------------------------------------------------
extern "C" void kernel_launch(void* const* d_in, const int* in_sizes, int n_in,
                              void* d_out, int out_size, void* d_ws, size_t ws_size,
                              hipStream_t stream) {
    const float* x  = (const float*)d_in[0];
    const float* Wc = (const float*)d_in[1];
    const float* bc = (const float*)d_in[2];
    const float* Wr = (const float*)d_in[3];
    const float* br = (const float*)d_in[4];
    float* out = (float*)d_out;

    unsigned short* Bhi = (unsigned short*)d_ws;
    unsigned short* Blo = Bhi + (size_t)NPAD * KDIM;
    float* bias = (float*)(Blo + (size_t)NPAD * KDIM);

    const int nRows = in_sizes[0] / KDIM;  // 8192

    const size_t total = (size_t)NPAD * KDIM;
    const int prepBlocks = (int)((total + 255) / 256);
    hipLaunchKernelGGL(prep_weights, dim3(prepBlocks), dim3(256), 0, stream,
                       Wc, bc, Wr, br, Bhi, Blo, bias);

    dim3 grid(NGRP, nRows / 16);
    hipLaunchKernelGGL(bbox_wmma_gemm, grid, dim3(32), 0, stream,
                       x, Bhi, Blo, bias, out, nRows);
}
